// CircuitSAT_15032385536527
// MI455X (gfx1250) — matrix operand order repaired and checked by reading, not compiled, not run on hardware
//
#include <hip/hip_runtime.h>
#include <hip/hip_bf16.h>
#include <math.h>

// ---------------------------------------------------------------------------
// CDNA5 (gfx1250) GNN implementation.
// Dense per-node GEMMs go through v_wmma_f32_16x16x32_bf16 (bf16 A/B, fp32 C).
// Each wave computes a 32-row x 64-col strip (2 row tiles x 4 col tiles):
// per K-chunk, A is loaded/converted once per row tile and each B tile is
// loaded once and reused by both row tiles -> 8 WMMAs per 16 b128 loads.
// ---------------------------------------------------------------------------

typedef __attribute__((ext_vector_type(16))) __bf16 v16bf;
typedef __attribute__((ext_vector_type(8)))  float  v8f;

#define DIM        100
#define DIM_P      128   // padded K for h / messages (4 chunks of 32)
#define DIM_AGG    50
#define DIM_AGG_P  64    // padded K for MLP hidden (2 chunks of 32)
#define DIM3       300
#define DIM3_P     320   // padded GRU gate width (20 col tiles = 5 groups of 4)
#define DIM_CLS    30
#define DIM_CLS_P  64    // padded classifier hidden (4 col tiles = 1 group)
#define N_ROUNDS   20

// ---------------------------------------------------------------------------
// Pack a fp32 weight matrix (Kreal x Nreal, row-major) into the WMMA B-operand
// layout: dst[((ntile*kc_count + kc)*32 + lane)*16 + e] where
//   col = ntile*16 + (lane & 15)
//   kk  = kc*32 + (lane>>4)*8 + (e&7) + (e>>3)*16   (matches A-operand K order)
// Out-of-range (padding) entries are zero.
// ---------------------------------------------------------------------------
__global__ void pack_b_kernel(const float* __restrict__ src, int kreal, int nreal,
                              int kc_count, int ntiles, __bf16* __restrict__ dst)
{
    int idx = blockIdx.x * blockDim.x + threadIdx.x;
    int total = ntiles * kc_count * 32 * 16;
    if (idx >= total) return;
    int e    = idx & 15;
    int lane = (idx >> 4) & 31;
    int kc   = (idx >> 9) % kc_count;
    int nt   = (idx >> 9) / kc_count;
    int col  = nt * 16 + (lane & 15);
    int kk   = kc * 32 + (lane >> 4) * 8 + (e & 7) + (e >> 3) * 16;
    float v = 0.0f;
    if (kk < kreal && col < nreal) v = src[(size_t)kk * nreal + col];
    dst[idx] = (__bf16)v;
}

// Load one 16x32 fp32 A tile row-slice for this lane and convert to the
// 16-bit A-matrix VGPR layout (ISA 7.12.2): elements 0..7 -> K = kb+0..7,
// elements 8..15 -> K = kb+16+0..7  (kb = kc*32 + half*8).
__device__ __forceinline__ v16bf load_a_tile(const float* __restrict__ arow, int kb)
{
    const float4* p0 = (const float4*)(arow + kb);
    const float4* p1 = (const float4*)(arow + kb + 16);
    float4 f0 = p0[0], f1 = p0[1];
    float4 f2 = p1[0], f3 = p1[1];
    v16bf a;
    a[0]  = (__bf16)f0.x;  a[1]  = (__bf16)f0.y;
    a[2]  = (__bf16)f0.z;  a[3]  = (__bf16)f0.w;
    a[4]  = (__bf16)f1.x;  a[5]  = (__bf16)f1.y;
    a[6]  = (__bf16)f1.z;  a[7]  = (__bf16)f1.w;
    a[8]  = (__bf16)f2.x;  a[9]  = (__bf16)f2.y;
    a[10] = (__bf16)f2.z;  a[11] = (__bf16)f2.w;
    a[12] = (__bf16)f3.x;  a[13] = (__bf16)f3.y;
    a[14] = (__bf16)f3.z;  a[15] = (__bf16)f3.w;
    return a;
}

// ---------------------------------------------------------------------------
// WMMA GEMM, 2 row tiles x 4 column tiles per wave:
//   C[32 x 64 strip] = A[32 x 32*kc_count] * Bpacked + bias (+ optional ReLU)
// grid = (ceil(M/32), colTiles/4), block = 32 (one wave).
// ---------------------------------------------------------------------------
__global__ void wmma_gemm_kernel(const float* __restrict__ A, int lda,
                                 const __bf16* __restrict__ Bp,
                                 const float* __restrict__ bias, int nreal,
                                 float* __restrict__ C, int ldc,
                                 int kc_count, int do_relu, int n_nodes)
{
    const int lane = threadIdx.x;          // 0..31
    const int rt   = blockIdx.x;           // 32-row strip
    const int ct0  = blockIdx.y * 4;       // first of 4 col tiles
    const int half = lane >> 4;            // 0 or 1
    const int r    = lane & 15;

    int row0 = rt * 32 + r;
    int row1 = rt * 32 + 16 + r;
    if (row0 >= n_nodes) row0 = n_nodes - 1;   // clamp (reads only)
    if (row1 >= n_nodes) row1 = n_nodes - 1;
    const float* __restrict__ arow0 = A + (size_t)row0 * lda;
    const float* __restrict__ arow1 = A + (size_t)row1 * lda;

    v8f acc[2][4];
#pragma unroll
    for (int rr = 0; rr < 2; ++rr)
#pragma unroll
        for (int j = 0; j < 4; ++j)
            acc[rr][j] = (v8f){0.f,0.f,0.f,0.f,0.f,0.f,0.f,0.f};

    const size_t tile_stride = (size_t)kc_count * 512;  // elements per col tile

    for (int kc = 0; kc < kc_count; ++kc) {
        const int kb = kc * 32 + half * 8;
        v16bf a0 = load_a_tile(arow0, kb);
        v16bf a1 = load_a_tile(arow1, kb);

        const __bf16* bk = Bp + ((size_t)ct0 * kc_count + kc) * 512 + (size_t)lane * 16;
#pragma unroll
        for (int j = 0; j < 4; ++j) {
            v16bf b = *(const v16bf*)(bk + (size_t)j * tile_stride);
            acc[0][j] = __builtin_amdgcn_wmma_f32_16x16x32_bf16(
                            false, a0, false, b, (short)0, acc[0][j], false, false);
            acc[1][j] = __builtin_amdgcn_wmma_f32_16x16x32_bf16(
                            false, a1, false, b, (short)0, acc[1][j], false, false);
        }
    }

    // C/D layout: VGPR i, lanes 0-15 -> M = i, N = lane; lanes 16-31 -> M = 8+i.
#pragma unroll
    for (int rr = 0; rr < 2; ++rr) {
        const int rbase = rt * 32 + rr * 16 + half * 8;
#pragma unroll
        for (int j = 0; j < 4; ++j) {
            const int col = (ct0 + j) * 16 + r;
            const float bv = (col < nreal) ? bias[col] : 0.0f;
            float* __restrict__ cp = C + (size_t)rbase * ldc + col;
#pragma unroll
            for (int i = 0; i < 8; ++i) {
                if (rbase + i < n_nodes) {
                    float v = acc[rr][j][i] + bv;
                    if (do_relu) v = fmaxf(v, 0.0f);
                    cp[(size_t)i * ldc] = v;
                }
            }
        }
    }
}

// ---------------------------------------------------------------------------
// h0 = features @ init_w + init_b   (K=4, trivial), zero-pads cols 100..127.
// ---------------------------------------------------------------------------
__global__ void init_h_kernel(const float* __restrict__ feat,
                              const float* __restrict__ iw,
                              const float* __restrict__ ib,
                              float* __restrict__ h, int n_nodes)
{
    int tid = blockIdx.x * blockDim.x + threadIdx.x;
    int n = tid >> 7;
    int d = tid & 127;
    if (n >= n_nodes) return;
    float v = 0.0f;
    if (d < DIM) {
        v = ib[d];
#pragma unroll
        for (int i = 0; i < 4; ++i)
            v += feat[(size_t)n * 4 + i] * iw[i * DIM + d];
    }
    h[(size_t)n * DIM_P + d] = v;
}

__global__ void fill_zero_kernel(float* __restrict__ p, size_t count)
{
    size_t tid = (size_t)blockIdx.x * blockDim.x + threadIdx.x;
    if (tid < count) p[tid] = 0.0f;
}

// ---------------------------------------------------------------------------
// SpMM scatter: dst[dst_idx[e]] += src[src_idx[e]] over all 128 padded cols.
// 32 lanes per edge, float4 per lane.
// ---------------------------------------------------------------------------
__global__ void spmm_kernel(const int* __restrict__ src_idx,
                            const int* __restrict__ dst_idx,
                            const float* __restrict__ src,
                            float* __restrict__ dst, int n_edges)
{
    int tid = blockIdx.x * blockDim.x + threadIdx.x;
    int e    = tid >> 5;
    int part = tid & 31;
    if (e >= n_edges) return;
    int c = src_idx[e];
    int r = dst_idx[e];
    const float4 v = *(const float4*)(src + (size_t)c * DIM_P + part * 4);
    float* d = dst + (size_t)r * DIM_P + part * 4;
    atomicAdd(d + 0, v.x);
    atomicAdd(d + 1, v.y);
    atomicAdd(d + 2, v.z);
    atomicAdd(d + 3, v.w);
}

// ---------------------------------------------------------------------------
// GRU gating: h = (1-z)*n + z*h   with torch gate order (r, z, n).
// gi/gh have row stride DIM3_P.  Writes only cols 0..99 of h.
// ---------------------------------------------------------------------------
__global__ void gru_gate_kernel(const float* __restrict__ gi,
                                const float* __restrict__ gh,
                                float* __restrict__ h, int n_nodes)
{
    int tid = blockIdx.x * blockDim.x + threadIdx.x;
    int n = tid / DIM;
    int d = tid % DIM;
    if (n >= n_nodes) return;
    const float* gir = gi + (size_t)n * DIM3_P;
    const float* ghr = gh + (size_t)n * DIM3_P;
    float ir  = gir[d], iz = gir[d + DIM], in = gir[d + 2 * DIM];
    float hr  = ghr[d], hz = ghr[d + DIM], hn = ghr[d + 2 * DIM];
    float rg = 1.0f / (1.0f + __expf(-(ir + hr)));
    float zg = 1.0f / (1.0f + __expf(-(iz + hz)));
    float ng = tanhf(in + rg * hn);
    float* hp = h + (size_t)n * DIM_P + d;
    float hv = *hp;
    *hp = (1.0f - zg) * ng + zg * hv;
}

// ---------------------------------------------------------------------------
// Classifier layer 2: out[n] = b2 + sum_j tmp[n][j] * w2[j]   (30 -> 1)
// ---------------------------------------------------------------------------
__global__ void cls2_kernel(const float* __restrict__ tmp,
                            const float* __restrict__ w2,
                            const float* __restrict__ b2,
                            float* __restrict__ out, int n_nodes)
{
    int n = blockIdx.x * blockDim.x + threadIdx.x;
    if (n >= n_nodes) return;
    float s = b2[0];
#pragma unroll
    for (int j = 0; j < DIM_CLS; ++j)
        s += tmp[(size_t)n * DIM_CLS_P + j] * w2[j];
    out[n] = s;
}

// ---------------------------------------------------------------------------
// Host-side orchestration.
// ---------------------------------------------------------------------------
static inline size_t align256(size_t x) { return (x + 255) & ~(size_t)255; }

extern "C" void kernel_launch(void* const* d_in, const int* in_sizes, int n_in,
                              void* d_out, int out_size, void* d_ws, size_t ws_size,
                              hipStream_t stream)
{
    (void)n_in; (void)out_size; (void)ws_size;

    const float* features = (const float*)d_in[0];
    const int*   edge_row = (const int*)d_in[1];
    const int*   edge_col = (const int*)d_in[2];
    const float* init_w   = (const float*)d_in[3];
    const float* init_b   = (const float*)d_in[4];
    const float* fm_w1 = (const float*)d_in[5];   const float* fm_b1 = (const float*)d_in[6];
    const float* fm_w2 = (const float*)d_in[7];   const float* fm_b2 = (const float*)d_in[8];
    const float* bm_w1 = (const float*)d_in[9];   const float* bm_b1 = (const float*)d_in[10];
    const float* bm_w2 = (const float*)d_in[11];  const float* bm_b2 = (const float*)d_in[12];
    const float* fg_wih = (const float*)d_in[13]; const float* fg_whh = (const float*)d_in[14];
    const float* fg_bih = (const float*)d_in[15]; const float* fg_bhh = (const float*)d_in[16];
    const float* bg_wih = (const float*)d_in[17]; const float* bg_whh = (const float*)d_in[18];
    const float* bg_bih = (const float*)d_in[19]; const float* bg_bhh = (const float*)d_in[20];
    const float* cl_w1 = (const float*)d_in[21];  const float* cl_b1 = (const float*)d_in[22];
    const float* cl_w2 = (const float*)d_in[23];  const float* cl_b2 = (const float*)d_in[24];

    const int N = in_sizes[0] / 4;   // nodes
    const int E = in_sizes[1];       // edges
    float* out = (float*)d_out;

    // ---- workspace carve-out --------------------------------------------
    char* ws = (char*)d_ws;
    size_t off = 0;
    auto carve = [&](size_t bytes) -> char* {
        char* p = ws + off;
        off = align256(off + bytes);
        return p;
    };
    float* h   = (float*)carve((size_t)N * DIM_P    * sizeof(float));
    float* a1  = (float*)carve((size_t)N * DIM_AGG_P * sizeof(float));
    float* m0  = (float*)carve((size_t)N * DIM_P    * sizeof(float));
    float* m1  = (float*)carve((size_t)N * DIM_P    * sizeof(float));
    float* gi  = (float*)carve((size_t)N * DIM3_P   * sizeof(float));
    float* gh  = (float*)carve((size_t)N * DIM3_P   * sizeof(float));
    float* ctmp= (float*)carve((size_t)N * DIM_CLS_P * sizeof(float));

    // packed bf16 weights: size = ntiles * kc * 512 elements
    auto carve_bp = [&](int ntiles, int kc) -> __bf16* {
        return (__bf16*)carve((size_t)ntiles * kc * 512 * sizeof(__bf16));
    };
    __bf16* fm_w1p = carve_bp(4, 4);    // 100x50  -> K128, N64   (4 tiles)
    __bf16* fm_w2p = carve_bp(8, 2);    // 50x100  -> K64,  N128  (8 tiles)
    __bf16* bm_w1p = carve_bp(4, 4);
    __bf16* bm_w2p = carve_bp(8, 2);
    __bf16* fg_wihp = carve_bp(20, 4);  // 100x300 -> K128, N320  (20 tiles)
    __bf16* fg_whhp = carve_bp(20, 4);
    __bf16* bg_wihp = carve_bp(20, 4);
    __bf16* bg_whhp = carve_bp(20, 4);
    __bf16* cl_w1p  = carve_bp(4, 4);   // 100x30  -> K128, N64   (4 tiles)

    // ---- pack weights ----------------------------------------------------
    auto pack = [&](const float* src, int kreal, int nreal, int kc, int nt, __bf16* dst) {
        int total = nt * kc * 512;
        hipLaunchKernelGGL(pack_b_kernel, dim3((total + 255) / 256), dim3(256), 0, stream,
                           src, kreal, nreal, kc, nt, dst);
    };
    pack(fm_w1, DIM, DIM_AGG, 4, 4, fm_w1p);
    pack(fm_w2, DIM_AGG, DIM, 2, 8, fm_w2p);
    pack(bm_w1, DIM, DIM_AGG, 4, 4, bm_w1p);
    pack(bm_w2, DIM_AGG, DIM, 2, 8, bm_w2p);
    pack(fg_wih, DIM, DIM3, 4, 20, fg_wihp);
    pack(fg_whh, DIM, DIM3, 4, 20, fg_whhp);
    pack(bg_wih, DIM, DIM3, 4, 20, bg_wihp);
    pack(bg_whh, DIM, DIM3, 4, 20, bg_whhp);
    pack(cl_w1, DIM, DIM_CLS, 4, 4, cl_w1p);

    // ---- init h ----------------------------------------------------------
    {
        size_t total = (size_t)N * DIM_P;
        hipLaunchKernelGGL(init_h_kernel, dim3((unsigned)((total + 255) / 256)), dim3(256),
                           0, stream, features, init_w, init_b, h, N);
    }

    const int rowStrips = (N + 31) / 32;
    auto gemm = [&](const float* A, int lda, const __bf16* Bp, const float* bias,
                    int nreal, float* C, int ldc, int kc, int colTiles, int relu) {
        hipLaunchKernelGGL(wmma_gemm_kernel, dim3(rowStrips, colTiles / 4), dim3(32),
                           0, stream, A, lda, Bp, bias, nreal, C, ldc, kc, relu, N);
    };

    auto half_round = [&](const __bf16* w1p, const float* b1,
                          const __bf16* w2p, const float* b2,
                          const int* sidx, const int* didx,
                          const __bf16* wihp, const float* bih,
                          const __bf16* whhp, const float* bhh) {
        // a1 = relu(h @ w1 + b1)         [N, 64]  (cols 50..63 zero)
        gemm(h, DIM_P, w1p, b1, DIM_AGG, a1, DIM_AGG_P, 4, 4, 1);
        // m0 = a1 @ w2 + b2              [N, 128] (cols 100..127 zero)
        gemm(a1, DIM_AGG_P, w2p, b2, DIM, m0, DIM_P, 2, 8, 0);
        // m1 = spmm(m0)
        {
            size_t total = (size_t)N * DIM_P;
            hipLaunchKernelGGL(fill_zero_kernel, dim3((unsigned)((total + 255) / 256)),
                               dim3(256), 0, stream, m1, total);
        }
        {
            size_t total = (size_t)E * 32;
            hipLaunchKernelGGL(spmm_kernel, dim3((unsigned)((total + 255) / 256)),
                               dim3(256), 0, stream, sidx, didx, m0, m1, E);
        }
        // gi = m1 @ wih + bih            [N, 320] (cols 300..319 zero)
        gemm(m1, DIM_P, wihp, bih, DIM3, gi, DIM3_P, 4, 20, 0);
        // gh = h @ whh + bhh             [N, 320]
        gemm(h, DIM_P, whhp, bhh, DIM3, gh, DIM3_P, 4, 20, 0);
        // h = GRU gate(gi, gh, h)
        {
            size_t total = (size_t)N * DIM;
            hipLaunchKernelGGL(gru_gate_kernel, dim3((unsigned)((total + 255) / 256)),
                               dim3(256), 0, stream, gi, gh, h, N);
        }
    };

    for (int round = 0; round < N_ROUNDS; ++round) {
        // forward: dst[row] += src[col]
        half_round(fm_w1p, fm_b1, fm_w2p, fm_b2, edge_col, edge_row,
                   fg_wihp, fg_bih, fg_whhp, fg_bhh);
        // backward: dst[col] += src[row]
        half_round(bm_w1p, bm_b1, bm_w2p, bm_b2, edge_row, edge_col,
                   bg_wihp, bg_bih, bg_whhp, bg_bhh);
    }

    // ---- classifier ------------------------------------------------------
    gemm(h, DIM_P, cl_w1p, cl_b1, DIM_CLS, ctmp, DIM_CLS_P, 4, 4, 1);
    hipLaunchKernelGGL(cls2_kernel, dim3((N + 255) / 256), dim3(256), 0, stream,
                       ctmp, cl_w2, cl_b2, out, N);
}
